// MultiHeadAttention_17179869497
// MI455X (gfx1250) — compile-verified, matrix operand order checked by reading, not run on hardware
//
#include <hip/hip_runtime.h>

// MI455X / gfx1250 fused multi-head attention:
//   1) qkv_proj_kernel : Q/K/V = X @ W^T   (f32 -> f16, WMMA f32_16x16x32_f16)
//   2) flash_attn_kernel: online-softmax attention; K tiles double-buffered
//      through the Tensor Data Mover (DMA overlaps WMMA compute)
//   3) out_proj_kernel : out = A @ Wo^T + bo; A tiles TDM double-buffered

typedef __attribute__((ext_vector_type(16))) _Float16 v16h;
typedef __attribute__((ext_vector_type(8)))  _Float16 v8h;
typedef __attribute__((ext_vector_type(8)))  float    v8f;
typedef __attribute__((ext_vector_type(4)))  float    v4f;
typedef __attribute__((ext_vector_type(4)))  unsigned int u32x4;
typedef __attribute__((ext_vector_type(8)))  int      i32x8;
typedef __attribute__((ext_vector_type(4)))  int      i32x4;

#define B_  4
#define S_  2048
#define D_  512
#define H_  8
#define DK_ 64

#if defined(__has_builtin)
#if __has_builtin(__builtin_amdgcn_tensor_load_to_lds) && \
    __has_builtin(__builtin_amdgcn_s_wait_tensorcnt)
#define HAVE_TDM 1
#endif
#endif
#ifndef HAVE_TDM
#define HAVE_TDM 0
#endif

__device__ __forceinline__ v8f wmma16(v16h a, v16h b, v8f c) {
  // (neg_a, A, neg_b, B, c_mod, C, reuse_a, reuse_b)
  return __builtin_amdgcn_wmma_f32_16x16x32_f16(false, a, false, b, (short)0, c,
                                                false, false);
}

// Build a 16-wide f16 A/B fragment for V_WMMA_*_16X16X32 from a row-major row.
// Per ISA layout: lanes 0-15 hold K = {0..7, 16..23}; lanes 16-31 K = {8..15, 24..31}.
__device__ __forceinline__ v16h frag16(const _Float16* p, int koff) {
  v8h lo = *(const v8h*)(p + koff);
  v8h hi = *(const v8h*)(p + koff + 16);
  v16h r;
  #pragma unroll
  for (int i = 0; i < 8; ++i) { r[i] = lo[i]; r[i + 8] = hi[i]; }
  return r;
}

#if HAVE_TDM
// Tensor Data Mover: DMA a 2D f16 tile (tile_w x tile_h, row-major, row stride
// `stride_elems` in memory) into LDS at byte offset `lds_off`, adding
// `pad_amount` DWORDs of LDS padding every `pad_interval` DWORDs so the tile
// lands with a bank-conflict-free padded row stride.
// D# packing per CDNA5 ISA (2D, groups 2/3 zero).
// 6-arg builtin: (u32x4 g0, i32x8 g1, i32x4 g2, i32x4 g3, i32x8, i32 cpol)
__device__ __forceinline__ void tdm_load_2d_f16(const _Float16* gsrc,
                                                unsigned int lds_off,
                                                int tile_w, int tile_h,
                                                int stride_elems,
                                                int pad_interval_code,
                                                int pad_amount_code) {
  unsigned long long ga = (unsigned long long)(uintptr_t)gsrc;
  u32x4 g0;
  g0[0] = 1u;                                        // count=1, user descriptor
  g0[1] = lds_off;                                   // lds_addr (bytes)
  g0[2] = (unsigned int)(ga & 0xffffffffull);        // global_addr[31:0]
  g0[3] = (unsigned int)((ga >> 32) & 0x1ffffffull)  // global_addr[56:32]
        | (2u << 30);                                // type = 2 ("image")
  i32x8 g1;
  g1[0] = (1 << 16)                                  // data_size: 2 bytes
        | (1 << 20)                                  // pad_enable
        | (pad_interval_code << 22)                  // pad_interval
        | (pad_amount_code << 25);                   // pad_amount
  g1[1] = (tile_w & 0xffff) << 16;                   // tensor_dim0[15:0]
  g1[2] = ((tile_w >> 16) & 0xffff)                  // tensor_dim0[31:16]
        | ((tile_h & 0xffff) << 16);                 // tensor_dim1[15:0]
  g1[3] = ((tile_h >> 16) & 0xffff)                  // tensor_dim1[31:16]
        | ((tile_w & 0xffff) << 16);                 // tile_dim0
  g1[4] = tile_h & 0xffff;                           // tile_dim1 (tile_dim2=0)
  g1[5] = stride_elems;                              // tensor_dim0_stride[31:0]
  g1[6] = 0;
  g1[7] = 0;
  i32x4 z4;
  z4[0] = 0; z4[1] = 0; z4[2] = 0; z4[3] = 0;
  i32x8 z8;
  #pragma unroll
  for (int i = 0; i < 8; ++i) z8[i] = 0;
  __builtin_amdgcn_tensor_load_to_lds(g0, g1, z4, z4, z8, 0);
}
#endif

// ---------------------------------------------------------------------------
// Kernel 1: fused Q/K/V projections.  Y[m,n] = sum_k X[m,k] * W[n,k]
// Output layout: [B, H, S, DK] f16.  Q is pre-scaled by 1/sqrt(DK).
// ---------------------------------------------------------------------------
__global__ __launch_bounds__(256)
void qkv_proj_kernel(const float* __restrict__ q, const float* __restrict__ k,
                     const float* __restrict__ v, const float* __restrict__ Wq,
                     const float* __restrict__ Wk, const float* __restrict__ Wv,
                     _Float16* __restrict__ Qh, _Float16* __restrict__ Kh,
                     _Float16* __restrict__ Vh) {
  __shared__ __align__(16) _Float16 sX[128 * 40];
  __shared__ __align__(16) _Float16 sW[64 * 40];

  const int tid = threadIdx.x;
  const int m0 = blockIdx.x * 128;
  const int n0 = blockIdx.y * 64;
  const int z  = blockIdx.z;
  const float* X = (z == 0) ? q  : (z == 1) ? k  : v;
  const float* W = (z == 0) ? Wq : (z == 1) ? Wk : Wv;
  _Float16*   Y  = (z == 0) ? Qh : (z == 1) ? Kh : Vh;
  const float scale = (z == 0) ? 0.125f : 1.0f;  // 1/sqrt(64) folded into Q

  const int wv = tid >> 5, lane = tid & 31, lm = lane & 15;
  const int koff = (lane < 16) ? 0 : 8;
  const int rowhalf = (lane >> 4) * 8;

  v8f acc[4];
  #pragma unroll
  for (int j = 0; j < 4; ++j)
    #pragma unroll
    for (int r = 0; r < 8; ++r) acc[j][r] = 0.f;

  for (int k0 = 0; k0 < D_; k0 += 32) {
    // stage X tile (128x32) f32->f16
    #pragma unroll
    for (int i = 0; i < 2; ++i) {
      int c = tid + i * 256;
      int row = c >> 2, c8 = (c & 3) * 8;
      const float* src = X + (size_t)(m0 + row) * D_ + k0 + c8;
      v4f f0 = *(const v4f*)src;
      v4f f1 = *(const v4f*)(src + 4);
      v8h hh;
      #pragma unroll
      for (int e = 0; e < 4; ++e) { hh[e] = (_Float16)f0[e]; hh[e + 4] = (_Float16)f1[e]; }
      *(v8h*)&sX[row * 40 + c8] = hh;
    }
    // stage W tile (64x32) f32->f16
    {
      int row = tid >> 2, c8 = (tid & 3) * 8;
      const float* src = W + (size_t)(n0 + row) * D_ + k0 + c8;
      v4f f0 = *(const v4f*)src;
      v4f f1 = *(const v4f*)(src + 4);
      v8h hh;
      #pragma unroll
      for (int e = 0; e < 4; ++e) { hh[e] = (_Float16)f0[e]; hh[e + 4] = (_Float16)f1[e]; }
      *(v8h*)&sW[row * 40 + c8] = hh;
    }
    __syncthreads();

    v16h a = frag16(&sX[(wv * 16 + lm) * 40], koff);
    #pragma unroll
    for (int j = 0; j < 4; ++j) {
      v16h bb = frag16(&sW[(j * 16 + lm) * 40], koff);
      acc[j] = wmma16(a, bb, acc[j]);
    }
    __syncthreads();
  }

  // scatter into [B, H, S, DK] f16
  #pragma unroll
  for (int j = 0; j < 4; ++j) {
    #pragma unroll
    for (int r = 0; r < 8; ++r) {
      int m = m0 + wv * 16 + rowhalf + r;
      int n = n0 + j * 16 + lm;
      int bb = m >> 11, s = m & 2047;
      int hh = n >> 6, dk = n & 63;
      Y[(((size_t)bb * H_ + hh) * S_ + s) * DK_ + dk] = (_Float16)(acc[j][r] * scale);
    }
  }
}

// ---------------------------------------------------------------------------
// Kernel 2: flash attention.  One block per (128 query rows, b*H + h).
// K tiles are double-buffered through the TDM: the DMA for tile kt+1 runs
// while tile kt's WMMAs and softmax execute.
// ---------------------------------------------------------------------------
__global__ __launch_bounds__(256)
void flash_attn_kernel(const _Float16* __restrict__ Qh,
                       const _Float16* __restrict__ Kh,
                       const _Float16* __restrict__ Vh,
                       const unsigned char* __restrict__ kpm,
                       _Float16* __restrict__ Ah) {
  __shared__ __align__(16) _Float16 sK[2][64 * 72];     // [buf][key][dk]
  __shared__ __align__(16) _Float16 sVt[64 * 72];       // [dk][key] (transposed)
  __shared__ __align__(16) _Float16 sP[8 * 16 * 72];    // per-wave P strips
  __shared__ float sMask[64];

  const int tid = threadIdx.x;
  const int q0 = blockIdx.x * 128;
  const int bh = blockIdx.y;            // b*H + h
  const int b = bh >> 3, h = bh & 7;
  const int wv = tid >> 5, lane = tid & 31, lm = lane & 15;
  const int koff = (lane < 16) ? 0 : 8;
  const int rowhalf = (lane >> 4) * 8;
  const int NT = S_ / 64;

  const _Float16* Qb = Qh + (size_t)bh * S_ * DK_;
  const _Float16* Kb = Kh + (size_t)bh * S_ * DK_;
  const _Float16* Vb = Vh + (size_t)bh * S_ * DK_;

  // Q fragments (16 rows x 64 dk) stay in registers for the whole key sweep
  v16h aq0, aq1;
  {
    const _Float16* qp = Qb + (size_t)(q0 + wv * 16 + lm) * DK_;
    aq0 = frag16(qp, koff);
    aq1 = frag16(qp + 32, koff);
  }

  float rm[8], rl[8];
  v8f oacc[4];
  #pragma unroll
  for (int r = 0; r < 8; ++r) { rm[r] = -1e30f; rl[r] = 0.f; }
  #pragma unroll
  for (int f = 0; f < 4; ++f)
    #pragma unroll
    for (int r = 0; r < 8; ++r) oacc[f][r] = 0.f;

#if HAVE_TDM
  const unsigned int sK_off0 = (unsigned int)(uintptr_t)(void*)&sK[0][0];
  const unsigned int sK_off1 = (unsigned int)(uintptr_t)(void*)&sK[1][0];
  // prologue: DMA tile 0 into buffer 0 (row stride 64 -> padded LDS stride 72)
  if (wv == 0)
    tdm_load_2d_f16(Kb, sK_off0, 64, 64, DK_, /*interval 32dw*/4, /*pad 4dw*/3);
#endif

  for (int kt = 0; kt < NT; ++kt) {
    const int cur = kt & 1;
    const _Float16* Ksrc = Kb + (size_t)kt * 64 * DK_;
    const _Float16* Vsrc = Vb + (size_t)kt * 64 * DK_;

    // V tile, transposed into LDS by the VALU path (TDM cannot transpose)
    #pragma unroll
    for (int i = 0; i < 2; ++i) {
      int c = tid + i * 256;
      int row = c >> 3, c8 = (c & 7) * 8;
#if !HAVE_TDM
      *(v8h*)&sK[cur][row * 72 + c8] = *(const v8h*)(Ksrc + row * DK_ + c8);
#endif
      v8h vvv = *(const v8h*)(Vsrc + row * DK_ + c8);
      #pragma unroll
      for (int e = 0; e < 8; ++e) sVt[(c8 + e) * 72 + row] = vvv[e];
      if (kt + 1 < NT)
        __builtin_prefetch((const void*)(Vsrc + 64 * DK_ + row * DK_ + c8), 0, 1);
    }
    if (tid < 64)
      sMask[tid] = kpm[(size_t)b * S_ + kt * 64 + tid] ? -1e30f : 0.0f;

#if HAVE_TDM
    if (wv == 0) __builtin_amdgcn_s_wait_tensorcnt(0);  // buffer `cur` landed
#endif
    __syncthreads();

#if HAVE_TDM
    // kick the DMA for the next K tile into the other buffer; it overlaps all
    // of this tile's WMMA + softmax work (everyone finished reading that
    // buffer before the previous iteration's trailing barrier).
    if (wv == 0 && kt + 1 < NT)
      tdm_load_2d_f16(Ksrc + 64 * DK_, cur ? sK_off0 : sK_off1, 64, 64, DK_, 4, 3);
#endif

    // scores S = Q K^T  (Q already carries 1/sqrt(dk))
    const _Float16* kbuf = &sK[cur][0];
    v8f sc[4];
    #pragma unroll
    for (int j = 0; j < 4; ++j) {
      #pragma unroll
      for (int r = 0; r < 8; ++r) sc[j][r] = 0.f;
      {
        v16h bk0 = frag16(kbuf + (j * 16 + lm) * 72, koff);
        sc[j] = wmma16(aq0, bk0, sc[j]);
        v16h bk1 = frag16(kbuf + (j * 16 + lm) * 72 + 32, koff);
        sc[j] = wmma16(aq1, bk1, sc[j]);
      }
      float mv = sMask[j * 16 + lm];
      #pragma unroll
      for (int r = 0; r < 8; ++r) sc[j][r] += mv;
    }

    // online softmax; row M = r + 8*(lane/16) lives across the 16-lane half
    #pragma unroll
    for (int r = 0; r < 8; ++r) {
      float mx = fmaxf(fmaxf(sc[0][r], sc[1][r]), fmaxf(sc[2][r], sc[3][r]));
      #pragma unroll
      for (int d = 1; d < 16; d <<= 1) mx = fmaxf(mx, __shfl_xor(mx, d, 32));
      float mnew = fmaxf(rm[r], mx);
      float fac = __expf(rm[r] - mnew);
      rm[r] = mnew;
      float ps = 0.f;
      #pragma unroll
      for (int j = 0; j < 4; ++j) {
        float p = __expf(sc[j][r] - mnew);
        sc[j][r] = p;
        ps += p;
      }
      #pragma unroll
      for (int d = 1; d < 16; d <<= 1) ps += __shfl_xor(ps, d, 32);
      rl[r] = rl[r] * fac + ps;
      #pragma unroll
      for (int f = 0; f < 4; ++f) oacc[f][r] *= fac;
    }

    // P: C/D layout -> LDS (per-wave strip) -> A-fragment layout (in-wave, no barrier)
    _Float16* pb = &sP[wv * 16 * 72];
    #pragma unroll
    for (int j = 0; j < 4; ++j)
      #pragma unroll
      for (int r = 0; r < 8; ++r)
        pb[(rowhalf + r) * 72 + j * 16 + lm] = (_Float16)sc[j][r];

    v16h ap0 = frag16(pb + lm * 72, koff);
    v16h ap1 = frag16(pb + lm * 72 + 32, koff);

    // O += P @ V  (V transposed in LDS -> contiguous B-fragment reads)
    #pragma unroll
    for (int f = 0; f < 4; ++f) {
      v16h bv0 = frag16(&sVt[(f * 16 + lm) * 72], koff);
      oacc[f] = wmma16(ap0, bv0, oacc[f]);
      v16h bv1 = frag16(&sVt[(f * 16 + lm) * 72 + 32], koff);
      oacc[f] = wmma16(ap1, bv1, oacc[f]);
    }
    __syncthreads();  // protects sVt/sMask before next iteration's loads
  }

  // epilogue: normalize, store [B, S, D] f16 with col = h*64 + dk
  #pragma unroll
  for (int f = 0; f < 4; ++f) {
    #pragma unroll
    for (int r = 0; r < 8; ++r) {
      int qrow = q0 + wv * 16 + rowhalf + r;
      float val = (rm[r] <= -1e29f) ? 0.f : oacc[f][r] / rl[r];  // fully-masked row -> 0
      Ah[((size_t)b * S_ + qrow) * D_ + h * DK_ + f * 16 + lm] = (_Float16)val;
    }
  }
}

// ---------------------------------------------------------------------------
// Kernel 3: output projection  out = A @ Wo^T + bo  (f32 out)
// A tiles (already f16 in memory) TDM double-buffered through padded LDS.
// ---------------------------------------------------------------------------
__global__ __launch_bounds__(256)
void out_proj_kernel(const _Float16* __restrict__ A, const float* __restrict__ Wo,
                     const float* __restrict__ bo, float* __restrict__ out) {
  __shared__ __align__(16) _Float16 sA[2][128 * 40];
  __shared__ __align__(16) _Float16 sW[64 * 40];

  const int tid = threadIdx.x;
  const int m0 = blockIdx.x * 128, n0 = blockIdx.y * 64;
  const int wv = tid >> 5, lane = tid & 31, lm = lane & 15;
  const int koff = (lane < 16) ? 0 : 8;
  const int rowhalf = (lane >> 4) * 8;

#if HAVE_TDM
  const unsigned int sA_off0 = (unsigned int)(uintptr_t)(void*)&sA[0][0];
  const unsigned int sA_off1 = (unsigned int)(uintptr_t)(void*)&sA[1][0];
  // prologue: DMA first A tile (32 wide x 128 high, mem row stride 512,
  // LDS row stride 40 via pad 4 DWORDs every 16 DWORDs)
  if (wv == 0)
    tdm_load_2d_f16(A + (size_t)m0 * D_, sA_off0, 32, 128, D_, 3, 3);
#endif

  v8f acc[4];
  #pragma unroll
  for (int j = 0; j < 4; ++j)
    #pragma unroll
    for (int r = 0; r < 8; ++r) acc[j][r] = 0.f;

  for (int k0 = 0; k0 < D_; k0 += 32) {
    const int cur = (k0 >> 5) & 1;
#if !HAVE_TDM
    #pragma unroll
    for (int i = 0; i < 2; ++i) {
      int c = tid + i * 256;
      int row = c >> 2, c8 = (c & 3) * 8;
      *(v8h*)&sA[cur][row * 40 + c8] =
          *(const v8h*)(A + (size_t)(m0 + row) * D_ + k0 + c8);
    }
#endif
    {
      int row = tid >> 2, c8 = (tid & 3) * 8;
      const float* src = Wo + (size_t)(n0 + row) * D_ + k0 + c8;
      v4f f0 = *(const v4f*)src;
      v4f f1 = *(const v4f*)(src + 4);
      v8h hh;
      #pragma unroll
      for (int e = 0; e < 4; ++e) { hh[e] = (_Float16)f0[e]; hh[e + 4] = (_Float16)f1[e]; }
      *(v8h*)&sW[row * 40 + c8] = hh;
    }
#if HAVE_TDM
    if (wv == 0) __builtin_amdgcn_s_wait_tensorcnt(0);
#endif
    __syncthreads();
#if HAVE_TDM
    if (wv == 0 && k0 + 32 < D_)
      tdm_load_2d_f16(A + (size_t)m0 * D_ + k0 + 32,
                      cur ? sA_off0 : sA_off1, 32, 128, D_, 3, 3);
#endif

    v16h a = frag16(&sA[cur][(wv * 16 + lm) * 40], koff);
    #pragma unroll
    for (int j = 0; j < 4; ++j) {
      v16h bb = frag16(&sW[(j * 16 + lm) * 40], koff);
      acc[j] = wmma16(a, bb, acc[j]);
    }
    __syncthreads();  // protects sW before next iteration's staging
  }

  #pragma unroll
  for (int j = 0; j < 4; ++j) {
    #pragma unroll
    for (int r = 0; r < 8; ++r) {
      int m = m0 + wv * 16 + rowhalf + r;
      int n = n0 + j * 16 + lm;
      out[(size_t)m * D_ + n] = acc[j][r] + bo[n];
    }
  }
}

// ---------------------------------------------------------------------------
extern "C" void kernel_launch(void* const* d_in, const int* in_sizes, int n_in,
                              void* d_out, int out_size, void* d_ws, size_t ws_size,
                              hipStream_t stream) {
  (void)in_sizes; (void)n_in; (void)out_size; (void)ws_size;
  const float* q  = (const float*)d_in[0];
  const float* k  = (const float*)d_in[1];
  const float* v  = (const float*)d_in[2];
  const unsigned char* kpm = (const unsigned char*)d_in[3];  // bool mask
  const float* Wq = (const float*)d_in[4];
  const float* Wk = (const float*)d_in[5];
  const float* Wv = (const float*)d_in[6];
  const float* Wo = (const float*)d_in[7];
  const float* bo = (const float*)d_in[8];
  float* out = (float*)d_out;

  const size_t NE = (size_t)B_ * S_ * D_;  // 4,194,304 elements
  _Float16* Qh = (_Float16*)d_ws;
  _Float16* Kh = Qh + NE;
  _Float16* Vh = Kh + NE;
  _Float16* Ah = Vh + NE;

  dim3 blk(256);
  dim3 g1((B_ * S_) / 128, D_ / 64, 3);
  qkv_proj_kernel<<<g1, blk, 0, stream>>>(q, k, v, Wq, Wk, Wv, Qh, Kh, Vh);

  dim3 g2(S_ / 128, B_ * H_);
  flash_attn_kernel<<<g2, blk, 0, stream>>>(Qh, Kh, Vh, kpm, Ah);

  dim3 g3((B_ * S_) / 128, D_ / 64);
  out_proj_kernel<<<g3, blk, 0, stream>>>(Ah, Wo, bo, out);
}
